// Diff2D3D_SAL_40312563040966
// MI455X (gfx1250) — matrix operand order repaired and checked by reading, not compiled, not run on hardware
//
#include <hip/hip_runtime.h>

// ---- problem constants (match reference) ----
#define M_IMG   1008
#define N_PCD   1024
#define CDIM    256
#define NI_F    76800
#define NP_F    30000
#define KI      64
#define NUM_CORR 256
#define CAND_CAP 4096
#define NPAIRS  (NUM_CORR * KI)   // 16384

typedef __attribute__((ext_vector_type(16))) _Float16 v16h;
typedef __attribute__((ext_vector_type(8)))  float    v8f;

// Build a 16-half WMMA fragment from four float4's (f32 -> f16 convert).
__device__ __forceinline__ v16h cvt16(float4 a0, float4 a1, float4 a2, float4 a3) {
  v16h r;
  r[0]=(_Float16)a0.x;  r[1]=(_Float16)a0.y;  r[2]=(_Float16)a0.z;  r[3]=(_Float16)a0.w;
  r[4]=(_Float16)a1.x;  r[5]=(_Float16)a1.y;  r[6]=(_Float16)a1.z;  r[7]=(_Float16)a1.w;
  r[8]=(_Float16)a2.x;  r[9]=(_Float16)a2.y;  r[10]=(_Float16)a2.z; r[11]=(_Float16)a2.w;
  r[12]=(_Float16)a3.x; r[13]=(_Float16)a3.y; r[14]=(_Float16)a3.z; r[15]=(_Float16)a3.w;
  return r;
}

// ---------------------------------------------------------------------------
// 1) Coarse GEMM: sim[M_IMG][N_PCD] = img_c @ pcd_c^T via v_wmma_f32_16x16x32_f16
//    One wave per 16x16 output tile. K = 256 in 8 chunks of 32.
// ---------------------------------------------------------------------------
__global__ __launch_bounds__(256) void coarse_gemm(const float* __restrict__ imgc,
                                                   const float* __restrict__ pcdc,
                                                   float* __restrict__ sim) {
  const int lane = threadIdx.x & 31;
  const int wave = threadIdx.x >> 5;
  const int tile = blockIdx.x * 8 + wave;      // 63*64 = 4032 tiles, 504 blocks
  const int tm = tile >> 6;                    // tile / 64
  const int tn = tile & 63;
  const int r15 = lane & 15;
  const int koA = (lane < 16) ? 0 : 16;        // A: lanes 0-15 K 0..15, 16-31 K 16..31
  const int koB = (lane < 16) ? 0 : 16;        // B: lanes 0-15 K 0..15, 16-31 K 16..31
  const float* arow = imgc + (tm * 16 + r15) * CDIM;
  const float* brow = pcdc + (tn * 16 + r15) * CDIM;
  v8f acc = {};
  #pragma unroll
  for (int kb = 0; kb < CDIM; kb += 32) {
    const float4* pa = (const float4*)(arow + kb);
    float4 a0 = pa[(koA >> 2)],     a1 = pa[(koA >> 2) + 1];
    float4 a2 = pa[2 + (koA >> 2)], a3 = pa[3 + (koA >> 2)];
    const float4* pb = (const float4*)(brow + kb + koB);
    float4 b0 = pb[0], b1 = pb[1], b2 = pb[2], b3 = pb[3];
    v16h af = cvt16(a0, a1, a2, a3);
    v16h bf = cvt16(b0, b1, b2, b3);
    acc = __builtin_amdgcn_wmma_f32_16x16x32_f16(false, af, false, bf,
                                                 (short)0, acc, false, false);
  }
  const int nb = tn * 16 + r15;
  const int mb = tm * 16 + ((lane < 16) ? 0 : 8);
  #pragma unroll
  for (int i = 0; i < 8; ++i) sim[(mb + i) * N_PCD + nb] = acc[i];
}

// ---------------------------------------------------------------------------
// 2) Row / column 3rd-largest (TOPK = 3)
// ---------------------------------------------------------------------------
__device__ __forceinline__ void ins3(float x, float& a0, float& a1, float& a2) {
  if (x > a0)      { a2 = a1; a1 = a0; a0 = x; }
  else if (x > a1) { a2 = a1; a1 = x; }
  else if (x > a2) { a2 = x; }
}

__global__ __launch_bounds__(256) void row_top3(const float* __restrict__ sim,
                                                const unsigned char* __restrict__ pmsk,
                                                float* __restrict__ rk) {
  const int row  = blockIdx.x * 8 + (threadIdx.x >> 5);   // 126 blocks
  const int lane = threadIdx.x & 31;
  if (row >= M_IMG) return;
  const float NEG = -3.0e38f;
  float t0 = NEG, t1 = NEG, t2 = NEG;
  for (int n = lane; n < N_PCD; n += 32) {
    float v = pmsk[n] ? sim[row * N_PCD + n] : NEG;
    ins3(v, t0, t1, t2);
  }
  #pragma unroll
  for (int off = 16; off; off >>= 1) {
    float b0 = __shfl_xor(t0, off, 32);
    float b1 = __shfl_xor(t1, off, 32);
    float b2 = __shfl_xor(t2, off, 32);
    ins3(b0, t0, t1, t2); ins3(b1, t0, t1, t2); ins3(b2, t0, t1, t2);
  }
  if (lane == 0) rk[row] = t2;
}

__global__ __launch_bounds__(256) void col_top3(const float* __restrict__ sim,
                                                const unsigned char* __restrict__ imsk,
                                                float* __restrict__ ck) {
  const int n = blockIdx.x * 256 + threadIdx.x;           // 4 blocks
  if (n >= N_PCD) return;
  const float NEG = -3.0e38f;
  float t0 = NEG, t1 = NEG, t2 = NEG;
  for (int m = 0; m < M_IMG; ++m) {
    float v = imsk[m] ? sim[m * N_PCD + n] : NEG;
    ins3(v, t0, t1, t2);
  }
  ck[n] = t2;
}

// ---------------------------------------------------------------------------
// 3) Mutual-top3 candidate collection (packed u64: ord(sim)<<32 | ~flatIdx)
// ---------------------------------------------------------------------------
__global__ __launch_bounds__(256) void cand_kernel(const float* __restrict__ sim,
                                                   const float* __restrict__ rk,
                                                   const float* __restrict__ ck,
                                                   const unsigned char* __restrict__ imsk,
                                                   const unsigned char* __restrict__ pmsk,
                                                   unsigned long long* __restrict__ cand,
                                                   unsigned* __restrict__ counter) {
  const int idx = blockIdx.x * 256 + threadIdx.x;         // 4032 blocks = M*N exactly
  const int m = idx >> 10, n = idx & 1023;
  const float v = sim[idx];
  if (imsk[m] && pmsk[n] && v > 0.0f && v >= rk[m] && v >= ck[n]) {
    unsigned pos = atomicAdd(counter, 1u);
    if (pos < CAND_CAP) {
      unsigned uo = __float_as_uint(v);
      uo = (uo & 0x80000000u) ? ~uo : (uo | 0x80000000u);  // monotone f32->u32
      cand[pos] = ((unsigned long long)uo << 32) | (unsigned)(~(unsigned)idx);
    }
  }
}

// ---------------------------------------------------------------------------
// 4) Global top-256 of candidates: single-block bitonic (descending)
// ---------------------------------------------------------------------------
__global__ __launch_bounds__(256) void top256_kernel(const unsigned long long* __restrict__ cand,
                                                     const unsigned* __restrict__ counter,
                                                     int* __restrict__ img_idx,
                                                     int* __restrict__ pcd_idx,
                                                     int* __restrict__ cvalid) {
  __shared__ unsigned long long sk[CAND_CAP];             // 32 KB
  const int tid = threadIdx.x;
  unsigned cnt = *counter; if (cnt > CAND_CAP) cnt = CAND_CAP;
  for (int i = tid; i < CAND_CAP; i += 256) sk[i] = (i < (int)cnt) ? cand[i] : 0ull;
  __syncthreads();
  for (int k = 2; k <= CAND_CAP; k <<= 1)
    for (int j = k >> 1; j > 0; j >>= 1) {
      for (int t = tid; t < CAND_CAP / 2; t += 256) {
        int i   = ((t & ~(j - 1)) << 1) | (t & (j - 1));
        int ixj = i | j;
        bool dir = ((i & k) == 0);
        unsigned long long a = sk[i], b = sk[ixj];
        if ((a < b) == dir) { sk[i] = b; sk[ixj] = a; }   // descending
      }
      __syncthreads();
    }
  if (tid < NUM_CORR) {
    bool v = tid < (int)cnt;
    unsigned flat = ~(unsigned)(sk[tid] & 0xffffffffull);
    img_idx[tid] = v ? (int)(flat >> 10)   : 0;
    pcd_idx[tid] = v ? (int)(flat & 1023)  : 0;
    cvalid[tid]  = v ? 1 : 0;
  }
}

// ---------------------------------------------------------------------------
// 5) Fine matching: one block per correspondence. 8 waves x 2 tiles cover the
//    64x64 sim via WMMA (gather rows through knn indices), then masked mutual
//    top-1 -> packed pair keys (img*32768 + pcd), sentinels sort last.
// ---------------------------------------------------------------------------
__global__ __launch_bounds__(256) void fine_match(const float* __restrict__ imgf,
                                                  const float* __restrict__ pcdf,
                                                  const int* __restrict__ knnI,
                                                  const int* __restrict__ knnP,
                                                  const unsigned char* __restrict__ kmsk,
                                                  const int* __restrict__ img_idx,
                                                  const int* __restrict__ pcd_idx,
                                                  const int* __restrict__ cvalid,
                                                  unsigned* __restrict__ pairs) {
  __shared__ float lsim[64 * 64];                         // 16 KB
  __shared__ int   knnIr[64], knnPr[64], mskr[64];
  __shared__ float rowVal[64];
  __shared__ int   rowBest[64], colBest[64];

  const int b = blockIdx.x, t = threadIdx.x;
  int mi = img_idx[b]; mi = (mi < 0) ? 0 : (mi > M_IMG - 1 ? M_IMG - 1 : mi);
  int pi = pcd_idx[b]; pi = (pi < 0) ? 0 : (pi > N_PCD - 1 ? N_PCD - 1 : pi);
  const int cv = cvalid[b];
  if (t < 64) {
    int ia = knnI[mi * KI + t]; ia = (ia < 0) ? 0 : (ia > NI_F - 1 ? NI_F - 1 : ia);
    int ib = knnP[pi * KI + t]; ib = (ib < 0) ? 0 : (ib > NP_F - 1 ? NP_F - 1 : ib);
    knnIr[t] = ia;
    knnPr[t] = ib;
    mskr[t]  = kmsk[pi * KI + t] ? 1 : 0;
  }
  __syncthreads();

  const int lane = t & 31, wid = t >> 5;
  const int r15 = lane & 15;
  const int koA = (lane < 16) ? 0 : 16;
  const int koB = (lane < 16) ? 0 : 16;
  #pragma unroll
  for (int tt = 0; tt < 2; ++tt) {
    const int tile = wid * 2 + tt;                        // 16 tiles of 16x16
    const int tm = tile >> 2, tn = tile & 3;
    const float* Ap = imgf + knnIr[tm * 16 + r15] * CDIM;
    const float* Bp = pcdf + knnPr[tn * 16 + r15] * CDIM;
    v8f acc = {};
    #pragma unroll
    for (int kb = 0; kb < CDIM; kb += 32) {
      const float4* pa = (const float4*)(Ap + kb);
      float4 a0 = pa[(koA >> 2)],     a1 = pa[(koA >> 2) + 1];
      float4 a2 = pa[2 + (koA >> 2)], a3 = pa[3 + (koA >> 2)];
      const float4* pb = (const float4*)(Bp + kb + koB);
      float4 b0 = pb[0], b1 = pb[1], b2 = pb[2], b3 = pb[3];
      v16h af = cvt16(a0, a1, a2, a3);
      v16h bf = cvt16(b0, b1, b2, b3);
      acc = __builtin_amdgcn_wmma_f32_16x16x32_f16(false, af, false, bf,
                                                   (short)0, acc, false, false);
    }
    const int nb = tn * 16 + r15;
    const int mb = tm * 16 + ((lane < 16) ? 0 : 8);
    #pragma unroll
    for (int i = 0; i < 8; ++i) lsim[(mb + i) * 64 + nb] = acc[i];
  }
  __syncthreads();

  const float NEG = -3.0e38f;
  if (t < 64) {                       // row max / argmax (first index on ties)
    float best = NEG; int bj = 0;
    for (int j = 0; j < 64; ++j)
      if (cv && mskr[j]) { float v = lsim[t * 64 + j]; if (v > best) { best = v; bj = j; } }
    rowVal[t] = best; rowBest[t] = bj;
  } else if (t < 128) {               // col argmax
    const int j = t - 64;
    float best = NEG; int br = 0;
    if (cv && mskr[j])
      for (int r = 0; r < 64; ++r) { float v = lsim[r * 64 + j]; if (v > best) { best = v; br = r; } }
    colBest[j] = br;
  }
  __syncthreads();

  if (t < 64) {
    const int bj = rowBest[t];
    const float bv = rowVal[t];
    const bool sel = (cv != 0) && (bv > 0.0f) && (colBest[bj] == t);
    unsigned ic = sel ? (unsigned)knnIr[t]  : (unsigned)NI_F;
    unsigned pc = sel ? (unsigned)knnPr[bj] : (unsigned)NP_F;
    pairs[b * KI + t] = ic * 32768u + pc;                 // fits in u32
  }
}

// ---------------------------------------------------------------------------
// 6) Lexsort of 16384 pair keys (ascending) -> replicates jnp.lexsort order
// ---------------------------------------------------------------------------
__global__ __launch_bounds__(1024) void sort_pairs(unsigned* __restrict__ pairs) {
  extern __shared__ unsigned sp[];                        // 64 KB dynamic
  const int tid = threadIdx.x;
  for (int i = tid; i < NPAIRS; i += 1024) sp[i] = pairs[i];
  __syncthreads();
  for (int k = 2; k <= NPAIRS; k <<= 1)
    for (int j = k >> 1; j > 0; j >>= 1) {
      for (int t = tid; t < NPAIRS / 2; t += 1024) {
        int i   = ((t & ~(j - 1)) << 1) | (t & (j - 1));
        int ixj = i | j;
        bool dir = ((i & k) == 0);
        unsigned a = sp[i], b = sp[ixj];
        if ((a > b) == dir) { sp[i] = b; sp[ixj] = a; }   // ascending
      }
      __syncthreads();
    }
  for (int i = tid; i < NPAIRS; i += 1024) pairs[i] = sp[i];
}

// ---------------------------------------------------------------------------
// 7) Dedup (first occurrence) + masked 256-dim dot per sorted pair (wave/pair)
// ---------------------------------------------------------------------------
__global__ __launch_bounds__(256) void score_kernel(const unsigned* __restrict__ keys,
                                                    const float* __restrict__ imgf,
                                                    const float* __restrict__ pcdf,
                                                    float* __restrict__ out) {
  const int w    = (blockIdx.x * 256 + threadIdx.x) >> 5; // global wave id
  const int lane = threadIdx.x & 31;
  if (w >= NPAIRS) return;
  const unsigned k = keys[w];
  const bool first = (w == 0) || (k != keys[w - 1]);
  const int img = (int)(k >> 15);
  const int pcd = (int)(k & 32767u);
  const bool valid = first && (img < NI_F);
  const int ic = (img > NI_F - 1) ? NI_F - 1 : img;
  const int pc = (pcd > NP_F - 1) ? NP_F - 1 : pcd;
  const float4* pi = (const float4*)(imgf + ic * CDIM) + lane * 2;
  const float4* pp = (const float4*)(pcdf + pc * CDIM) + lane * 2;
  float4 x0 = pi[0], x1 = pi[1], y0 = pp[0], y1 = pp[1];
  float s = x0.x * y0.x + x0.y * y0.y + x0.z * y0.z + x0.w * y0.w
          + x1.x * y1.x + x1.y * y1.y + x1.z * y1.z + x1.w * y1.w;
  #pragma unroll
  for (int off = 16; off; off >>= 1) s += __shfl_xor(s, off, 32);
  if (lane == 0) out[w] = valid ? s : 0.0f;
}

// ---------------------------------------------------------------------------
// Launch: all kernels on `stream`; workspace carved from d_ws (~4.3 MB used).
// ---------------------------------------------------------------------------
extern "C" void kernel_launch(void* const* d_in, const int* in_sizes, int n_in,
                              void* d_out, int out_size, void* d_ws, size_t ws_size,
                              hipStream_t stream) {
  const float*         imgc = (const float*)d_in[0];
  const float*         pcdc = (const float*)d_in[1];
  const float*         imgf = (const float*)d_in[2];
  const float*         pcdf = (const float*)d_in[3];
  const unsigned char* imsk = (const unsigned char*)d_in[4];
  const unsigned char* pmsk = (const unsigned char*)d_in[5];
  const int*           knnI = (const int*)d_in[6];
  const int*           knnP = (const int*)d_in[7];
  const unsigned char* kmsk = (const unsigned char*)d_in[8];
  float* out = (float*)d_out;

  char* ws = (char*)d_ws;
  float*              sim     = (float*)(ws + 0);                  // 4,128,768 B
  float*              row_kth = (float*)(ws + 4128768);            // 4 KB slot
  float*              col_kth = (float*)(ws + 4132864);            // 4 KB slot
  unsigned*           counter = (unsigned*)(ws + 4136960);         // 256 B slot
  unsigned long long* cand    = (unsigned long long*)(ws + 4137216); // 32 KB
  int*                img_idx = (int*)(ws + 4169984);
  int*                pcd_idx = (int*)(ws + 4171008);
  int*                cvalid  = (int*)(ws + 4172032);
  unsigned*           pairs   = (unsigned*)(ws + 4173056);         // 64 KB

  hipMemsetAsync(counter, 0, sizeof(unsigned), stream);

  coarse_gemm <<<504,  256, 0, stream>>>(imgc, pcdc, sim);
  row_top3    <<<126,  256, 0, stream>>>(sim, pmsk, row_kth);
  col_top3    <<<4,    256, 0, stream>>>(sim, imsk, col_kth);
  cand_kernel <<<4032, 256, 0, stream>>>(sim, row_kth, col_kth, imsk, pmsk, cand, counter);
  top256_kernel<<<1,   256, 0, stream>>>(cand, counter, img_idx, pcd_idx, cvalid);
  fine_match  <<<NUM_CORR, 256, 0, stream>>>(imgf, pcdf, knnI, knnP, kmsk,
                                             img_idx, pcd_idx, cvalid, pairs);
  sort_pairs  <<<1, 1024, NPAIRS * sizeof(unsigned), stream>>>(pairs);
  score_kernel<<<(NPAIRS * 32) / 256, 256, 0, stream>>>(pairs, imgf, pcdf, out);

  (void)in_sizes; (void)n_in; (void)out_size; (void)ws_size;
}
